// MultiHeadAttention_33732673143108
// MI455X (gfx1250) — compile-verified, matrix operand order checked by reading, not run on hardware
//
#include <hip/hip_runtime.h>

// ---------------------------------------------------------------------------
// MHA (B=2,C=1024,L=2048,H=16,D=64) + residual + LayerNorm for MI455X/gfx1250.
// Matrix math: V_WMMA_F32_16X16X32_BF16 (wave32 fragments).
// Weight staging: TDM tensor_load_to_lds (TENSORcnt) with cooperative fallback.
// ---------------------------------------------------------------------------

#define B_ 2
#define C_ 1024
#define L_ 2048
#define H_ 16
#define D_ 64
#define LN_EPS 1e-5f

typedef __attribute__((ext_vector_type(16))) __bf16 v16bf;
typedef __attribute__((ext_vector_type(8)))  float  v8f;
typedef __attribute__((ext_vector_type(4)))  unsigned int u32x4;
typedef __attribute__((ext_vector_type(8)))  int i32x8;
typedef __attribute__((ext_vector_type(4)))  int i32x4;

#if defined(__has_builtin)
#  if __has_builtin(__builtin_amdgcn_tensor_load_to_lds)
#    define HAVE_TDM 1
#  else
#    define HAVE_TDM 0
#  endif
#else
#  define HAVE_TDM 0
#endif

union Frag16 {
    v16bf v;
    u32x4 q[2];
};

// round-to-nearest-even f32 -> bf16 (bit pattern as ushort)
__device__ __forceinline__ unsigned short f32_to_bf16(float f) {
    unsigned int u = __float_as_uint(f);
    u += 0x7FFFu + ((u >> 16) & 1u);
    return (unsigned short)(u >> 16);
}

// Load one 16x32 bf16 A/B fragment half-row: elements 0..7 = K[base..base+7],
// elements 8..15 = K[base+16..base+23] (ISA 7.12.2 16-bit A/B layout).
// p must point at K=base for this lane; both chunks are 16B-aligned.
__device__ __forceinline__ v16bf load_frag16(const unsigned short* p) {
    Frag16 f;
    f.q[0] = *(const u32x4*)(p);        // K base .. base+7
    f.q[1] = *(const u32x4*)(p + 16);   // K base+16 .. base+23
    return f.v;
}

__device__ __forceinline__ v8f wmma_bf16(v16bf a, v16bf b, v8f c) {
    return __builtin_amdgcn_wmma_f32_16x16x32_bf16(false, a, false, b,
                                                   (short)0, c, false, false);
}

// ---------------------------------------------------------------------------
// Kernel 1: x[B,C,L] -> xT_bf16[B,L,C] (and optional xT_f32 for the residual)
// ---------------------------------------------------------------------------
__global__ __launch_bounds__(256) void convert_xT(const float* __restrict__ in,
                                                  unsigned short* __restrict__ out_bf,
                                                  float* __restrict__ out_f32) {
    size_t idx = (size_t)blockIdx.x * 256 + threadIdx.x;   // over B*L*C
    if (idx >= (size_t)B_ * L_ * C_) return;
    int c = (int)(idx % C_);
    int l = (int)((idx / C_) % L_);
    int b = (int)(idx / ((size_t)C_ * L_));
    float v = in[((size_t)b * C_ + c) * L_ + l];
    out_bf[idx] = f32_to_bf16(v);
    if (out_f32) out_f32[idx] = v;
}

// ---------------------------------------------------------------------------
// Kernel 2: W[H,C,D] -> Wt_bf16[H,D,C]  (column of W == contiguous row of Wt)
// ---------------------------------------------------------------------------
__global__ __launch_bounds__(256) void convert_w(const float* __restrict__ in,
                                                 unsigned short* __restrict__ out) {
    size_t idx = (size_t)blockIdx.x * 256 + threadIdx.x;   // over H*C*D
    if (idx >= (size_t)H_ * C_ * D_) return;
    int d = (int)(idx % D_);
    int c = (int)((idx / D_) % C_);
    int h = (int)(idx / ((size_t)C_ * D_));
    out[((size_t)h * D_ + d) * C_ + c] = f32_to_bf16(in[idx]);
}

// ---------------------------------------------------------------------------
// Kernel 3: QKV projection.  Out = A(LxC) @ Wt^T(CxD) + bias, per (b,h).
// Grid: (L/64, B*H). Block: 128 = 4 waves; wave w owns rows m0 = 64*bx + 16*w
// and computes the FULL D=64 (4 accumulators) so the A fragment is reused 4x.
// Wt[h] (64x1024 bf16 = 128 KB) is staged into LDS once per workgroup via the
// Tensor Data Mover (single 2D descriptor), waves read B-fragments from LDS.
// transposed==0: Out[B*H][L][D] bf16 (Q,K); ==1: Out[B*H][D][L] (V).
// ---------------------------------------------------------------------------
__global__ __launch_bounds__(128) void qkv_gemm(const unsigned short* __restrict__ A,
                                                const unsigned short* __restrict__ Wt,
                                                const float* __restrict__ bias,
                                                unsigned short* __restrict__ Out,
                                                int transposed) {
    extern __shared__ char smem[];
    unsigned short* lw = (unsigned short*)smem;       // [D_][C_] bf16 = 128 KB

    const int bh   = blockIdx.y;
    const int b    = bh / H_;
    const int h    = bh % H_;
    const int tid  = threadIdx.x;
    const int wave = tid >> 5;
    const int lane = tid & 31;
    const int hi   = lane >> 4;       // K-half select (ISA fragment layout)
    const int nn   = lane & 15;       // A: row, B: column
    const int m0   = blockIdx.x * 64 + wave * 16;

    const unsigned short* wsrc = Wt + (size_t)h * D_ * C_;

    // ---- Stage Wt[h] into LDS ----
#if HAVE_TDM
    if (wave == 0) {
        // TDM D# (ISA 8.3/8.4): 2D tile 64 rows x 1024 cols of 2-byte elems.
        unsigned long long ga = (unsigned long long)(uintptr_t)wsrc;
        unsigned int lds_off  = (unsigned int)(uintptr_t)lw; // flat addr[31:0] = LDS offset
        u32x4 g0;
        g0.x = 1u;                                            // count=1 (valid user D#)
        g0.y = lds_off;                                       // lds_addr
        g0.z = (unsigned int)(ga & 0xFFFFFFFFu);              // global_addr[31:0]
        g0.w = (unsigned int)((ga >> 32) & 0x01FFFFFFu)       // global_addr[56:32]
             | (2u << 30);                                    // type=2 ("image")
        i32x8 g1;
        g1[0] = 1 << 16;                                      // data_size=1 (2 bytes)
        g1[1] = (int)((C_ & 0xFFFF) << 16);                   // tensor_dim0[15:0]
        g1[2] = (int)(((C_ >> 16) & 0xFFFF) | ((D_ & 0xFFFF) << 16)); // dim0 hi | tensor_dim1 lo
        g1[3] = (int)(((D_ >> 16) & 0xFFFF) | ((C_ & 0xFFFF) << 16)); // dim1 hi | tile_dim0
        g1[4] = D_;                                           // tile_dim1 (tile_dim2=0)
        g1[5] = C_;                                           // tensor_dim0_stride[31:0]
        g1[6] = 0;                                            // stride hi | dim1_stride lo
        g1[7] = 0;
        i32x4 gz = {0, 0, 0, 0};
#if __clang_major__ >= 23
        i32x8 gz8 = {0, 0, 0, 0, 0, 0, 0, 0};
        __builtin_amdgcn_tensor_load_to_lds(g0, g1, gz, gz, gz8, 0);
#else
        __builtin_amdgcn_tensor_load_to_lds(g0, g1, gz, gz, 0);
#endif
        __builtin_amdgcn_s_wait_tensorcnt(0);
    }
#else
    // Cooperative fallback: 8192 x 16B chunks across 128 threads.
    {
        const u32x4* src = (const u32x4*)wsrc;
        u32x4*       dst = (u32x4*)lw;
        for (int i = tid; i < (D_ * C_) / 8; i += 128)
            dst[i] = src[i];
    }
#endif
    __syncthreads();

    // ---- GEMM: 16 rows x 64 cols per wave, K over C=1024 in steps of 32 ----
    v8f acc[4];
#pragma unroll
    for (int dt = 0; dt < 4; ++dt)
        acc[dt] = (v8f){0.f, 0.f, 0.f, 0.f, 0.f, 0.f, 0.f, 0.f};

    const unsigned short* arow = A  + ((size_t)(b * L_ + m0 + nn)) * C_ + 8 * hi;
    const unsigned short* brow = lw + (size_t)nn * C_ + 8 * hi;

    for (int k0 = 0; k0 < C_; k0 += 32) {
        v16bf a = load_frag16(arow + k0);
#pragma unroll
        for (int dt = 0; dt < 4; ++dt) {
            v16bf w = load_frag16(brow + (size_t)dt * 16 * C_ + k0);
            acc[dt] = wmma_bf16(a, w, acc[dt]);
        }
    }

#pragma unroll
    for (int dt = 0; dt < 4; ++dt) {
        const float bval = bias[h * D_ + dt * 16 + nn];
#pragma unroll
        for (int r = 0; r < 8; ++r) {
            int m = m0 + r + 8 * hi;               // C/D layout: M = r (+8 hi)
            unsigned short e = f32_to_bf16(acc[dt][r] + bval);
            if (transposed)
                Out[((size_t)(bh * D_ + dt * 16 + nn)) * L_ + m] = e;  // Vt[d][l]
            else
                Out[((size_t)(bh * L_ + m)) * D_ + dt * 16 + nn] = e;  // Q/K[l][d]
        }
    }
}

// ---------------------------------------------------------------------------
// Kernel 4: attention for one (b, h, 16-query-row tile).
// Grid: (L/16, B*H). Block: 256 = 8 waves.
// LDS: scores 16x2048 f32 (128KB) + P 16x2048 bf16 (64KB) + reductions.
// ---------------------------------------------------------------------------
__global__ __launch_bounds__(256) void attn_kernel(const unsigned short* __restrict__ Q,
                                                   const unsigned short* __restrict__ K,
                                                   const unsigned short* __restrict__ Vt,
                                                   float* __restrict__ att) {
    extern __shared__ char smem[];
    float*          sc     = (float*)smem;                   // [16][2048]
    float*          ctxred = sc + 16 * 2048;                 // [16][64]
    float*          red    = ctxred + 16 * 64;               // [16][16]
    float*          rowmax = red + 256;                      // [16]
    float*          rowsum = rowmax + 16;                    // [16]
    unsigned short* pbuf   = (unsigned short*)(rowsum + 16); // [16][2048] (16B-aligned)

    const int bh   = blockIdx.y;
    const int b    = bh / H_;
    const int h    = bh % H_;
    const int m0   = blockIdx.x * 16;
    const int tid  = threadIdx.x;
    const int wave = tid >> 5;
    const int lane = tid & 31;
    const int hi   = lane >> 4;
    const int nn   = lane & 15;

    for (int i = tid; i < 16 * 64; i += 256) ctxred[i] = 0.f;

    // Q fragments (A-matrix over K-dim = D = 64 -> two k-steps)
    const unsigned short* qrow = Q + ((size_t)(bh * L_ + m0 + nn)) * D_ + 8 * hi;
    v16bf qa0 = load_frag16(qrow);
    v16bf qa1 = load_frag16(qrow + 32);

    const float scale = 0.125f;  // 1/sqrt(64)

    // ---- Phase 1: scores ----
    for (int nt = wave; nt < L_ / 16; nt += 8) {
        const int n0 = nt * 16;
        const unsigned short* krow = K + ((size_t)(bh * L_ + n0 + nn)) * D_ + 8 * hi;
        v16bf kb0 = load_frag16(krow);
        v16bf kb1 = load_frag16(krow + 32);
        v8f acc = {0.f, 0.f, 0.f, 0.f, 0.f, 0.f, 0.f, 0.f};
        acc = wmma_bf16(qa0, kb0, acc);
        acc = wmma_bf16(qa1, kb1, acc);
#pragma unroll
        for (int r = 0; r < 8; ++r)
            sc[(r + 8 * hi) * 2048 + n0 + nn] = acc[r] * scale;
    }
    __syncthreads();

    // ---- Phase 2: softmax (16 rows x 16 segments of 128 columns) ----
    {
        const int row = tid >> 4;
        const int seg = tid & 15;
        float* srow = sc + row * 2048 + seg * 128;
        float m = -3.4e38f;
        for (int j = 0; j < 128; ++j) m = fmaxf(m, srow[j]);
        red[row * 16 + seg] = m;
        __syncthreads();
        if (seg == 0) {
            float mm = red[row * 16];
            for (int j = 1; j < 16; ++j) mm = fmaxf(mm, red[row * 16 + j]);
            rowmax[row] = mm;
        }
        __syncthreads();
        const float mx = rowmax[row];
        unsigned short* prow = pbuf + row * 2048 + seg * 128;
        float s = 0.f;
        for (int j = 0; j < 128; ++j) {
            float e = __expf(srow[j] - mx);
            prow[j] = f32_to_bf16(e);
            s += e;
        }
        red[row * 16 + seg] = s;
        __syncthreads();
        if (seg == 0) {
            float ss = 0.f;
            for (int j = 0; j < 16; ++j) ss += red[row * 16 + j];
            rowsum[row] = ss;
        }
        __syncthreads();
    }

    // ---- Phase 3: ctx = P @ V, K-dim (keys) split across 8 waves ----
    v8f acc4[4];
#pragma unroll
    for (int dt = 0; dt < 4; ++dt)
        acc4[dt] = (v8f){0.f, 0.f, 0.f, 0.f, 0.f, 0.f, 0.f, 0.f};

    for (int kt = wave; kt < L_ / 32; kt += 8) {
        v16bf pa = load_frag16(pbuf + nn * 2048 + kt * 32 + 8 * hi);
#pragma unroll
        for (int dt = 0; dt < 4; ++dt) {
            const unsigned short* vrow =
                Vt + ((size_t)(bh * D_ + dt * 16 + nn)) * L_ + kt * 32 + 8 * hi;
            v16bf vb = load_frag16(vrow);
            acc4[dt] = wmma_bf16(pa, vb, acc4[dt]);
        }
    }
#pragma unroll
    for (int dt = 0; dt < 4; ++dt)
#pragma unroll
        for (int r = 0; r < 8; ++r)
            atomicAdd(&ctxred[(r + 8 * hi) * 64 + dt * 16 + nn], acc4[dt][r]);
    __syncthreads();

    for (int e = tid; e < 16 * 64; e += 256) {
        int m = e >> 6, d = e & 63;
        att[((size_t)(b * L_ + m0 + m)) * (H_ * D_) + h * D_ + d] =
            ctxred[e] / rowsum[m];
    }
}

// ---------------------------------------------------------------------------
// Kernel 5: residual + LayerNorm over C=1024. Block per (b,l) row, 256 threads.
// ---------------------------------------------------------------------------
__global__ __launch_bounds__(256) void resid_ln(const float* __restrict__ xT,
                                                const float* __restrict__ att,
                                                const float* __restrict__ gamma,
                                                const float* __restrict__ beta,
                                                float* __restrict__ out) {
    __shared__ float r1[256];
    __shared__ float r2[256];
    const size_t row = blockIdx.x;
    const int tid = threadIdx.x;

    float hv[4];
    float s = 0.f, ss = 0.f;
#pragma unroll
    for (int i = 0; i < 4; ++i) {
        int c = tid + i * 256;
        float v = xT[row * C_ + c] + att[row * C_ + c];
        hv[i] = v;
        s += v;
        ss += v * v;
    }
    r1[tid] = s;
    r2[tid] = ss;
    __syncthreads();
    for (int o = 128; o > 0; o >>= 1) {
        if (tid < o) {
            r1[tid] += r1[tid + o];
            r2[tid] += r2[tid + o];
        }
        __syncthreads();
    }
    const float mu  = r1[0] * (1.f / C_);
    const float var = r2[0] * (1.f / C_) - mu * mu;
    const float rs  = rsqrtf(var + LN_EPS);
#pragma unroll
    for (int i = 0; i < 4; ++i) {
        int c = tid + i * 256;
        out[row * C_ + c] = (hv[i] - mu) * rs * gamma[c] + beta[c];
    }
}

// ---------------------------------------------------------------------------
// Host launcher
// ---------------------------------------------------------------------------
extern "C" void kernel_launch(void* const* d_in, const int* in_sizes, int n_in,
                              void* d_out, int out_size, void* d_ws, size_t ws_size,
                              hipStream_t stream) {
    (void)in_sizes; (void)n_in; (void)out_size; (void)ws_size;

    const float* x     = (const float*)d_in[0];
    const float* y     = (const float*)d_in[1];
    const float* Wq    = (const float*)d_in[2];
    const float* Wk    = (const float*)d_in[3];
    const float* Wv    = (const float*)d_in[4];
    const float* bq    = (const float*)d_in[5];
    const float* bk    = (const float*)d_in[6];
    const float* bv    = (const float*)d_in[7];
    const float* gamma = (const float*)d_in[8];
    const float* beta  = (const float*)d_in[9];
    float* out = (float*)d_out;

    // Workspace carve (~78 MB total)
    char* ws = (char*)d_ws;
    size_t off = 0;
    auto carve = [&](size_t bytes) -> void* {
        void* p = ws + off;
        off += (bytes + 255) & ~(size_t)255;
        return p;
    };
    const size_t NBLC = (size_t)B_ * L_ * C_;          // 4M elems
    const size_t NW   = (size_t)H_ * D_ * C_;          // 1M elems
    const size_t NQKV = (size_t)B_ * H_ * L_ * D_;     // 4M elems

    unsigned short* xTbf = (unsigned short*)carve(NBLC * 2);
    unsigned short* yTbf = (unsigned short*)carve(NBLC * 2);
    float*          xTf  = (float*)carve(NBLC * 4);
    unsigned short* Wqt  = (unsigned short*)carve(NW * 2);
    unsigned short* Wkt  = (unsigned short*)carve(NW * 2);
    unsigned short* Wvt  = (unsigned short*)carve(NW * 2);
    unsigned short* Qb   = (unsigned short*)carve(NQKV * 2);
    unsigned short* Kb   = (unsigned short*)carve(NQKV * 2);
    unsigned short* Vtb  = (unsigned short*)carve(NQKV * 2);
    float*          attb = (float*)carve(NBLC * 4);

    // 1) transpose + bf16 convert
    convert_xT<<<(int)(NBLC / 256), 256, 0, stream>>>(x, xTbf, xTf);
    convert_xT<<<(int)(NBLC / 256), 256, 0, stream>>>(y, yTbf, nullptr);
    const int nw = (int)((size_t)H_ * C_ * D_);
    convert_w<<<nw / 256, 256, 0, stream>>>(Wq, Wqt);
    convert_w<<<nw / 256, 256, 0, stream>>>(Wk, Wkt);
    convert_w<<<nw / 256, 256, 0, stream>>>(Wv, Wvt);

    // 2) QKV projections (WMMA, TDM-staged weights in 128 KB LDS)
    const size_t wsmem = (size_t)D_ * C_ * 2;   // 131072 B
    (void)hipFuncSetAttribute((const void*)qkv_gemm,
                              hipFuncAttributeMaxDynamicSharedMemorySize,
                              (int)wsmem);
    dim3 qgrid(L_ / 64, B_ * H_);
    qkv_gemm<<<qgrid, 128, wsmem, stream>>>(xTbf, Wqt, bq, Qb, 0);
    qkv_gemm<<<qgrid, 128, wsmem, stream>>>(yTbf, Wkt, bk, Kb, 0);
    qkv_gemm<<<qgrid, 128, wsmem, stream>>>(yTbf, Wvt, bv, Vtb, 1);

    // 3) attention (WMMA + LDS softmax); ~202 KB LDS of the 320 KB/WGP
    const size_t smem = (size_t)16 * 2048 * 4   // scores f32
                      + (size_t)16 * 64 * 4     // ctx reduction
                      + 256 * 4 + 16 * 4 + 16 * 4
                      + (size_t)16 * 2048 * 2;  // P bf16
    (void)hipFuncSetAttribute((const void*)attn_kernel,
                              hipFuncAttributeMaxDynamicSharedMemorySize,
                              (int)smem);
    dim3 agrid(L_ / 16, B_ * H_);
    attn_kernel<<<agrid, 256, smem, stream>>>(Qb, Kb, Vtb, attb);

    // 4) residual + LayerNorm
    resid_ln<<<B_ * L_, 256, 0, stream>>>(xTf, attb, gamma, beta, out);
}
